// NonLocalStack_25391846654138
// MI455X (gfx1250) — matrix operand order repaired and checked by reading, not compiled, not run on hardware
//
#include <hip/hip_runtime.h>
#include <hip/hip_bf16.h>
#include <math.h>

typedef _Float16 f16;
typedef __attribute__((ext_vector_type(16))) _Float16 v16h;
typedef __attribute__((ext_vector_type(8)))  float    v8f;
typedef __attribute__((ext_vector_type(4)))  unsigned int u32x4;
typedef __attribute__((ext_vector_type(8)))  int      i32x8;
typedef __attribute__((ext_vector_type(4)))  int      i32x4;

#define Tt 4
#define Hh 128
#define Ww 128
#define Cc 32
#define KNN 8
#define NVOX (Tt*Hh*Ww)      /* 65536 */
#define STKC (KNN*Cc)        /* 256   */

static __device__ __forceinline__ v8f wmma32(v16h a, v16h b, v8f c) {
  // D = A(16x32 f16) * B(32x16 f16) + C(16x16 f32)
  return __builtin_amdgcn_wmma_f32_16x16x32_f16(false, a, false, b, (short)0, c, false, false);
}

// A-fragment from an f16 matrix with row stride `ld`, in this code's contiguous k order (not AMD's table order):
// lane l holds row m=l%16; halfs 0..15 -> K=(l/16)*16+j (contiguous, matches b_frag)
static __device__ __forceinline__ v16h a_frag(const f16* base, int ld, int m0,
                                              int kbase, int lm, int hi) {
  const f16* p = base + (size_t)(m0 + lm) * ld + kbase;
  v16h a;
#pragma unroll
  for (int j = 0; j < 8; ++j) a[j] = p[hi * 16 + j];
#pragma unroll
  for (int j = 0; j < 8; ++j) a[8 + j] = p[hi * 16 + 8 + j];
  return a;
}

// B-fragment: lane l holds column n=l%16, 16 contiguous K starting at kbase+(l/16)*16,
// matrix stored transposed as [N][K] so this is one contiguous 32B read per lane.
static __device__ __forceinline__ v16h b_frag(const f16* p) {
  v16h b;
#pragma unroll
  for (int j = 0; j < 16; ++j) b[j] = p[j];
  return b;
}

// ---------------------------------------------------------------------------
// Kernel 1: weight pack/transpose f32 -> f16 [N][K] layouts
// ---------------------------------------------------------------------------
__global__ __launch_bounds__(256) void k_pack(
    const float* __restrict__ cw, const float* __restrict__ qw,
    const float* __restrict__ pw, const float* __restrict__ f1,
    const float* __restrict__ f2, f16* __restrict__ cwT, f16* __restrict__ qwT,
    f16* __restrict__ pwT, f16* __restrict__ f1T, f16* __restrict__ f2T) {
  int i = blockIdx.x * 256 + threadIdx.x;
  if (i < 32 * 9 * 256) {  // conv: [c_out][tap][ci] <- [c_out][ci][tap]
    int co = i / (9 * 256); int r = i % (9 * 256); int tap = r / 256; int ci = r % 256;
    cwT[i] = (f16)cw[(co * 256 + ci) * 9 + tap];
  }
  if (i < 2 * 96 * 32) {   // qkv: [blk][n=96][k=32] <- [blk][k][n]
    int blk = i / (96 * 32); int r = i % (96 * 32); int n = r / 32; int k = r % 32;
    qwT[i] = (f16)qw[(blk * 32 + k) * 96 + n];
  }
  if (i < 2 * 32 * 32) {   // proj: [blk][n=32][k=32]
    int blk = i / 1024; int r = i % 1024; int n = r / 32; int k = r % 32;
    pwT[i] = (f16)pw[(blk * 32 + k) * 32 + n];
  }
  if (i < 2 * 64 * 32) {   // fc1: [blk][n=64][k=32]
    int blk = i / 2048; int r = i % 2048; int n = r / 32; int k = r % 32;
    f1T[i] = (f16)f1[(blk * 32 + k) * 64 + n];
  }
  if (i < 2 * 32 * 64) {   // fc2: [blk][n=32][k=64]
    int blk = i / 2048; int r = i % 2048; int n = r / 64; int k = r % 64;
    f2T[i] = (f16)f2[(blk * 64 + k) * 32 + n];
  }
}

// ---------------------------------------------------------------------------
// Kernel 2: non-local kNN search + softmax-weighted gather stacking.
// One thread per voxel; output channels-last f16: stk[voxel][k*32+c].
// ---------------------------------------------------------------------------
__global__ __launch_bounds__(256) void k_nl_stack(const float* __restrict__ vid,
                                                  f16* __restrict__ stk) {
  int v = blockIdx.x * 256 + threadIdx.x;
  int t = v >> 14;
  int hw = v & 16383;
  int h = hw >> 7;
  int w = hw & 127;

  // warm L2/L0 for the next temporal plane (global_prefetch_b8)
  __builtin_prefetch(vid + (size_t)min(t + 1, Tt - 1) * (Hh * Ww) + hw, 0, 1);

  float cen[32];
#pragma unroll
  for (int c = 0; c < 32; ++c) cen[c] = vid[c * NVOX + v];

  float bd[8]; int bo[8];
#pragma unroll
  for (int j = 0; j < 8; ++j) { bd[j] = 3.4e38f; bo[j] = 0; }

  for (int o = 0; o < 243; ++o) {
    int dt = o / 81 - 1;
    int r = o % 81;
    int dh = r / 9 - 4;
    int dw = r % 9 - 4;
    int ti = min(max(t + dt, 0), Tt - 1);
    int hi = min(max(h + dh, 0), Hh - 1);
    int wi = min(max(w + dw, 0), Ww - 1);
    int nv = (ti * Hh + hi) * Ww + wi;
    float d = 0.f;
#pragma unroll
    for (int c = 0; c < 32; ++c) {
      float df = cen[c] - vid[c * NVOX + nv];
      d = fmaf(df, df, d);
    }
    if (d < bd[7]) {  // stable sorted insertion (ascending dist == top_k(-d))
      bd[7] = d; bo[7] = o;
#pragma unroll
      for (int j = 7; j > 0; --j) {
        if (bd[j] < bd[j - 1]) {
          float td = bd[j]; bd[j] = bd[j - 1]; bd[j - 1] = td;
          int to = bo[j]; bo[j] = bo[j - 1]; bo[j - 1] = to;
        }
      }
    }
  }
  // softmax over -dist (max is -bd[0])
  float wk[8]; float wsum = 0.f;
#pragma unroll
  for (int j = 0; j < 8; ++j) { wk[j] = __expf(bd[0] - bd[j]); wsum += wk[j]; }
  float inv = 1.f / wsum;
#pragma unroll
  for (int k = 0; k < 8; ++k) {
    int o = bo[k];
    int dt = o / 81 - 1;
    int r = o % 81;
    int dh = r / 9 - 4;
    int dw = r % 9 - 4;
    int ti = min(max(t + dt, 0), Tt - 1);
    int hi = min(max(h + dh, 0), Hh - 1);
    int wi = min(max(w + dw, 0), Ww - 1);
    int nv = (ti * Hh + hi) * Ww + wi;
    float wt = wk[k] * inv;
#pragma unroll
    for (int c = 0; c < 32; ++c)
      stk[(size_t)v * STKC + k * 32 + c] = (f16)(wt * vid[c * NVOX + nv]);
  }
}

// ---------------------------------------------------------------------------
// Kernel 3: conv (1,3,3) 256->32 as implicit GEMM via WMMA.
// Block = 256 thr (8 waves) computes 64 positions x 32 c_out; M=64,N=32,K=2304.
// The full packed weight (32 x 2304 f16 = 144KB) is DMA'd into LDS once per
// workgroup via the Tensor Data Mover; B-fragments then come from LDS.
// ---------------------------------------------------------------------------
__global__ __launch_bounds__(256) void k_conv(const f16* __restrict__ stk,
                                              const f16* __restrict__ cwT,
                                              const float* __restrict__ cb,
                                              float* __restrict__ xbuf) {
  HIP_DYNAMIC_SHARED(f16, wlds);      // 32*2304 f16 = 147456 B dynamic LDS

#if __has_builtin(__builtin_amdgcn_tensor_load_to_lds)
  if (threadIdx.x < 32) {             // one wave issues the TDM descriptor
    unsigned lds_base = (unsigned)(size_t)(void*)wlds;   // LDS aperture: addr[31:0]
    unsigned long long ga = (unsigned long long)(size_t)cwT;
    // D# group0: count=1 | lds_addr | global_addr[56:0] | type=2
    u32x4 g0;
    g0[0] = 1u;
    g0[1] = lds_base;
    g0[2] = (unsigned)(ga & 0xFFFFFFFFu);
    g0[3] = (unsigned)((ga >> 32) & 0x01FFFFFFu) | (2u << 30);
    // D# group1: 2D tensor, data_size=2B, dim0=2304, dim1=32, tile=2304x32,
    // dim0_stride=2304 (elements)
    i32x8 g1;
    g1[0] = 0x00010000;               // workgroup_mask=0, data_size=1 (2B)
    g1[1] = (int)(2304u << 16);       // tensor_dim0[15:0] in bits[63:48]
    g1[2] = (int)(32u << 16);         // tensor_dim0[31:16]=0 | tensor_dim1[15:0]
    g1[3] = (int)(2304u << 16);       // tensor_dim1[31:16]=0 | tile_dim0
    g1[4] = 32;                       // tile_dim1=32 | tile_dim2=0
    g1[5] = 2304;                     // tensor_dim0_stride[31:0]
    g1[6] = 0;                        // stride0[47:32]=0 | dim1_stride[15:0]=0
    g1[7] = 0;
    i32x4 z4 = {0, 0, 0, 0};          // groups 2/3 unused (<=2D tensor)
    i32x8 z8 = {0, 0, 0, 0, 0, 0, 0, 0};
    __builtin_amdgcn_tensor_load_to_lds(g0, g1, z4, z4, z8, 0);
    __builtin_amdgcn_s_wait_tensorcnt(0);
  }
  __syncthreads();
  const f16* wsrc = wlds;
#else
  const f16* wsrc = cwT;
#endif

  int row = blockIdx.x >> 1;          // t*128 + h
  int w0 = (blockIdx.x & 1) * 64;
  int t = row >> 7;
  int h = row & 127;
  int lane = threadIdx.x & 31;
  int wv = threadIdx.x >> 5;          // 0..7
  int mtile = wv >> 1;                // 0..3
  int ntile = wv & 1;                 // 0..1
  int lm = lane & 15;
  int hi = lane >> 4;

  v8f acc;
#pragma unroll
  for (int j = 0; j < 8; ++j) acc[j] = 0.f;

  for (int kh = 0; kh < 3; ++kh) {
    int hh = h + kh - 1;
    if (hh < 0 || hh >= Hh) continue;       // uniform per block (zero pad)
    for (int kw = 0; kw < 3; ++kw) {
      int dw = kw - 1;
      int tap = kh * 3 + kw;
      int wq = w0 + mtile * 16 + lm + dw;
      bool valid = (wq >= 0) && (wq < Ww);
      const f16* arow = stk + (size_t)((t * Hh + hh) * Ww + wq) * STKC;
      for (int kc = 0; kc < 8; ++kc) {
        v16h a;
#pragma unroll
        for (int j = 0; j < 16; ++j) a[j] = (f16)0.f;
        if (valid) {
          const f16* p = arow + kc * 32 + hi * 16;
#pragma unroll
          for (int j = 0; j < 8; ++j) a[j] = p[j];
#pragma unroll
          for (int j = 0; j < 8; ++j) a[8 + j] = p[8 + j];
        }
        const f16* bp =
            wsrc + (size_t)((ntile * 16 + lm) * 9 + tap) * 256 + kc * 32 + hi * 16;
        acc = wmma32(a, b_frag(bp), acc);
      }
    }
  }
  int n = ntile * 16 + lm;
  float bias = cb[n];
#pragma unroll
  for (int j = 0; j < 8; ++j) {
    int m = mtile * 16 + hi * 8 + j;
    int vox = row * Ww + w0 + m;
    xbuf[(size_t)vox * 32 + n] = acc[j] + bias;
  }
}

// ---------------------------------------------------------------------------
// Kernel 4: both transformer blocks on one 8x8 window (64 tokens x 32 ch),
// fully in LDS; epilogue fuses vid + out with the NCHW transpose.
// ---------------------------------------------------------------------------
__global__ __launch_bounds__(256) void k_xform(
    const float* __restrict__ xbuf, const float* __restrict__ vid,
    const f16* __restrict__ qwT, const float* __restrict__ qb,
    const f16* __restrict__ pwT, const float* __restrict__ pb,
    const f16* __restrict__ f1T, const float* __restrict__ f1b,
    const f16* __restrict__ f2T, const float* __restrict__ f2b,
    const float* __restrict__ ln1g, const float* __restrict__ ln1b,
    const float* __restrict__ ln2g, const float* __restrict__ ln2b,
    float* __restrict__ out) {
  __shared__ float xt[64][32];   // residual stream
  __shared__ f16 h16[64][64];    // f16 A-operand staging (LN out / attn out)
  __shared__ f16 g16[64][64];    // gelu(hidden) f16
  __shared__ float qkv[64][96];  // q|k|v

  int tid = threadIdx.x;
  int lane = tid & 31;
  int wv = tid >> 5;
  int lm = lane & 15;
  int hi = lane >> 4;

  int widx = blockIdx.x;
  int t = widx >> 8;
  int rem = widx & 255;
  int nh = rem >> 4;
  int nw = rem & 15;

#pragma unroll
  for (int i = 0; i < 8; ++i) {  // load 64x32 tile
    int e = tid * 8 + i;
    int l = e >> 5; int c = e & 31;
    int vox = (t * Hh + nh * 8 + (l >> 3)) * Ww + nw * 8 + (l & 7);
    xt[l][c] = xbuf[(size_t)vox * 32 + c];
  }
  __syncthreads();

  for (int blk = 0; blk < 2; ++blk) {
    // ---- LN1 -> h16 (f16) ----
    if (tid < 64) {
      float s = 0.f, s2 = 0.f;
#pragma unroll
      for (int c = 0; c < 32; ++c) { float x = xt[tid][c]; s += x; s2 += x * x; }
      float m = s * (1.f / 32.f);
      float rs = rsqrtf(s2 * (1.f / 32.f) - m * m + 1e-5f);
#pragma unroll
      for (int c = 0; c < 32; ++c)
        h16[tid][c] = (f16)((xt[tid][c] - m) * rs * ln1g[blk * 32 + c] + ln1b[blk * 32 + c]);
    }
    __syncthreads();

    // ---- qkv GEMM: M=64,N=96,K=32 -> 24 tiles, 3 per wave ----
#pragma unroll
    for (int i = 0; i < 3; ++i) {
      int tile = wv * 3 + i;
      int mt = tile / 6, nt = tile % 6;
      v8f c;
#pragma unroll
      for (int j = 0; j < 8; ++j) c[j] = 0.f;
      v16h a = a_frag(&h16[0][0], 64, mt * 16, 0, lm, hi);
      const f16* bp = qwT + (size_t)(blk * 96 + nt * 16 + lm) * 32 + hi * 16;
      c = wmma32(a, b_frag(bp), c);
      float bias = qb[blk * 96 + nt * 16 + lm];
#pragma unroll
      for (int j = 0; j < 8; ++j)
        qkv[mt * 16 + hi * 8 + j][nt * 16 + lm] = c[j] + bias;
    }
    __syncthreads();

    // ---- attention: 4 heads x 64 rows, head dim 8 (VALU, 2-pass softmax) ----
    {
      int head = tid >> 6;
      int qr = tid & 63;
      float qv[8];
#pragma unroll
      for (int j = 0; j < 8; ++j) qv[j] = qkv[qr][head * 8 + j];
      const float scale = 0.3535533905932738f;  // 8^-0.5
      float mx = -3.4e38f;
      for (int kr = 0; kr < 64; ++kr) {
        float d = 0.f;
#pragma unroll
        for (int j = 0; j < 8; ++j) d = fmaf(qv[j], qkv[kr][32 + head * 8 + j], d);
        mx = fmaxf(mx, d * scale);
      }
      float sum = 0.f; float ov[8];
#pragma unroll
      for (int j = 0; j < 8; ++j) ov[j] = 0.f;
      for (int kr = 0; kr < 64; ++kr) {
        float d = 0.f;
#pragma unroll
        for (int j = 0; j < 8; ++j) d = fmaf(qv[j], qkv[kr][32 + head * 8 + j], d);
        float p = __expf(d * scale - mx);
        sum += p;
#pragma unroll
        for (int j = 0; j < 8; ++j) ov[j] = fmaf(p, qkv[kr][64 + head * 8 + j], ov[j]);
      }
      float inv = 1.f / sum;
#pragma unroll
      for (int j = 0; j < 8; ++j) h16[qr][head * 8 + j] = (f16)(ov[j] * inv);
    }
    __syncthreads();

    // ---- proj GEMM: M=64,N=32,K=32 -> 8 tiles, 1 per wave; xt += result ----
    {
      int mt = wv >> 1, nt = wv & 1;
      v8f c;
#pragma unroll
      for (int j = 0; j < 8; ++j) c[j] = 0.f;
      v16h a = a_frag(&h16[0][0], 64, mt * 16, 0, lm, hi);
      const f16* bp = pwT + (size_t)(blk * 32 + nt * 16 + lm) * 32 + hi * 16;
      c = wmma32(a, b_frag(bp), c);
      float bias = pb[blk * 32 + nt * 16 + lm];
#pragma unroll
      for (int j = 0; j < 8; ++j)
        xt[mt * 16 + hi * 8 + j][nt * 16 + lm] += c[j] + bias;
    }
    __syncthreads();

    // ---- LN2 -> h16 ----
    if (tid < 64) {
      float s = 0.f, s2 = 0.f;
#pragma unroll
      for (int c = 0; c < 32; ++c) { float x = xt[tid][c]; s += x; s2 += x * x; }
      float m = s * (1.f / 32.f);
      float rs = rsqrtf(s2 * (1.f / 32.f) - m * m + 1e-5f);
#pragma unroll
      for (int c = 0; c < 32; ++c)
        h16[tid][c] = (f16)((xt[tid][c] - m) * rs * ln2g[blk * 32 + c] + ln2b[blk * 32 + c]);
    }
    __syncthreads();

    // ---- fc1 + gelu: M=64,N=64,K=32 -> 16 tiles, 2 per wave -> g16 ----
#pragma unroll
    for (int i = 0; i < 2; ++i) {
      int tile = wv * 2 + i;
      int mt = tile >> 2, nt = tile & 3;
      v8f c;
#pragma unroll
      for (int j = 0; j < 8; ++j) c[j] = 0.f;
      v16h a = a_frag(&h16[0][0], 64, mt * 16, 0, lm, hi);
      const f16* bp = f1T + (size_t)(blk * 64 + nt * 16 + lm) * 32 + hi * 16;
      c = wmma32(a, b_frag(bp), c);
      float bias = f1b[blk * 64 + nt * 16 + lm];
#pragma unroll
      for (int j = 0; j < 8; ++j) {
        float x = c[j] + bias;
        float g = 0.5f * x * (1.f + tanhf(0.7978845608f * (x + 0.044715f * x * x * x)));
        g16[mt * 16 + hi * 8 + j][nt * 16 + lm] = (f16)g;
      }
    }
    __syncthreads();

    // ---- fc2: M=64,N=32,K=64 -> 8 tiles, 1 per wave, 2 K-steps; xt += ----
    {
      int mt = wv >> 1, nt = wv & 1;
      v8f c;
#pragma unroll
      for (int j = 0; j < 8; ++j) c[j] = 0.f;
#pragma unroll
      for (int ks = 0; ks < 64; ks += 32) {
        v16h a = a_frag(&g16[0][0], 64, mt * 16, ks, lm, hi);
        const f16* bp = f2T + (size_t)(blk * 32 + nt * 16 + lm) * 64 + ks + hi * 16;
        c = wmma32(a, b_frag(bp), c);
      }
      float bias = f2b[blk * 32 + nt * 16 + lm];
#pragma unroll
      for (int j = 0; j < 8; ++j)
        xt[mt * 16 + hi * 8 + j][nt * 16 + lm] += c[j] + bias;
    }
    __syncthreads();
  }

  // ---- epilogue: out[c,t,h,w] = vid + xt (fused residual + transpose) ----
#pragma unroll
  for (int i = 0; i < 8; ++i) {
    int e = tid * 8 + i;
    int l = e >> 5; int c = e & 31;
    int gh = nh * 8 + (l >> 3), gw = nw * 8 + (l & 7);
    size_t oidx = (((size_t)c * Tt + t) * Hh + gh) * Ww + gw;
    out[oidx] = vid[oidx] + xt[l][c];
  }
}

// ---------------------------------------------------------------------------
extern "C" void kernel_launch(void* const* d_in, const int* in_sizes, int n_in,
                              void* d_out, int out_size, void* d_ws, size_t ws_size,
                              hipStream_t stream) {
  (void)in_sizes; (void)n_in; (void)out_size; (void)ws_size;
  const float* vid    = (const float*)d_in[0];
  const float* conv_w = (const float*)d_in[1];
  const float* conv_b = (const float*)d_in[2];
  const float* ln1_g  = (const float*)d_in[3];
  const float* ln1_b  = (const float*)d_in[4];
  const float* qkv_w  = (const float*)d_in[5];
  const float* qkv_b  = (const float*)d_in[6];
  const float* proj_w = (const float*)d_in[7];
  const float* proj_b = (const float*)d_in[8];
  const float* ln2_g  = (const float*)d_in[9];
  const float* ln2_b  = (const float*)d_in[10];
  const float* fc1_w  = (const float*)d_in[11];
  const float* fc1_b  = (const float*)d_in[12];
  const float* fc2_w  = (const float*)d_in[13];
  const float* fc2_b  = (const float*)d_in[14];

  char* ws = (char*)d_ws;
  f16*   stk  = (f16*)(ws);                       // 65536*256*2 = 33554432 B
  float* xbuf = (float*)(ws + 33554432);          // 65536*32*4  =  8388608 B
  f16*   cwT  = (f16*)(ws + 41943040);            // 147456 B
  f16*   qwT  = (f16*)(ws + 42090496);            // 12288 B
  f16*   pwT  = (f16*)(ws + 42102784);            // 4096 B
  f16*   f1T  = (f16*)(ws + 42106880);            // 8192 B
  f16*   f2T  = (f16*)(ws + 42115072);            // 8192 B

  k_pack<<<288, 256, 0, stream>>>(conv_w, qkv_w, proj_w, fc1_w, fc2_w,
                                  cwT, qwT, pwT, f1T, f2T);
  k_nl_stack<<<NVOX / 256, 256, 0, stream>>>(vid, stk);
  k_conv<<<(Tt * Hh) * (Ww / 64), 256, 147456, stream>>>(stk, cwT, conv_b, xbuf);
  k_xform<<<Tt * 16 * 16, 256, 0, stream>>>(xbuf, vid,
                                            qwT, qkv_b, pwT, proj_b,
                                            f1T, fc1_b, f2T, fc2_b,
                                            ln1_g, ln1_b, ln2_g, ln2_b,
                                            (float*)d_out);
}
